// CrossAttention_541165879819
// MI455X (gfx1250) — matrix operand order repaired and checked by reading, not compiled, not run on hardware
//
#include <hip/hip_runtime.h>
#include <hip/hip_bf16.h>
#include <math.h>

typedef __attribute__((ext_vector_type(16))) _Float16 v16h;
typedef __attribute__((ext_vector_type(8)))  _Float16 v8h;
typedef __attribute__((ext_vector_type(8)))  float    v8f;

#define DIMX 512
#define HIDX 512
#define NH   8
#define HD   64
#define BB   2
#define NQ   1024
#define NK   2048

#define CAT16(lo, hi) __builtin_shufflevector(lo, hi, 0, 1, 2, 3, 4, 5, 6, 7, 8, 9, 10, 11, 12, 13, 14, 15)

__device__ __forceinline__ v8f wmma_f16(v16h a, v16h b, v8f c) {
  // D = A(16x32 f16) * B(32x16 f16) + C(16x16 f32)
  return __builtin_amdgcn_wmma_f32_16x16x32_f16(false, a, false, b, (short)0, c, false, false);
}

__device__ __forceinline__ float fast_rcp(float x) { return __builtin_amdgcn_rcpf(x); }
__device__ __forceinline__ float fast_silu(float x) {
  return x * fast_rcp(1.0f + __expf(-x));
}

// ---- async global->LDS copy (CDNA5 GLOBAL_LOAD_ASYNC_TO_LDS_B128, ASYNCcnt) ----
#if __has_builtin(__builtin_amdgcn_global_load_async_to_lds_b128)
#define USE_ASYNC_LDS 1
#else
#define USE_ASYNC_LDS 0
#endif

typedef int intv4 __attribute__((vector_size(16)));
typedef __attribute__((address_space(1))) intv4 gv4;   // global int4
typedef __attribute__((address_space(3))) intv4 lv4;   // LDS int4

__device__ __forceinline__ void cp_async_b128(const void* g, void* l) {
#if USE_ASYNC_LDS
  __builtin_amdgcn_global_load_async_to_lds_b128((gv4*)g, (lv4*)l, 0, 0);
#else
  *(v8h*)l = *(const v8h*)g;
#endif
}
__device__ __forceinline__ void cp_async_wait() {
#if USE_ASYNC_LDS
  asm volatile("s_wait_asynccnt 0x0" ::: "memory");
#endif
}

// ---------------------------------------------------------------- conversions
__global__ void cvt_act(const float* __restrict__ in, _Float16* __restrict__ out, int n) {
  int i = blockIdx.x * 256 + threadIdx.x;
  if (i < n) out[i] = (_Float16)in[i];
}

// WT[n][k] = W[k][n] * scale   (512x512)
__global__ void cvt_wT(const float* __restrict__ W, _Float16* __restrict__ WT, float scale) {
  int i = blockIdx.x * 256 + threadIdx.x;
  int n = i >> 9, k = i & 511;
  WT[n * 512 + k] = (_Float16)(W[k * 512 + n] * scale);
}

// ---------------------------------------------------------------- WMMA GEMM
// C(MxN=512) = A(Mx512) * BT^T, BT is N x K row-major (pre-transposed weights).
// MODE 0: f32 row-major (ld=512). MODE 1: f16 [B][H][seqN][64]. MODE 2: f16 [B][H][64][seqN].
template <int MODE>
__global__ __launch_bounds__(128) void gemm_f16k(const _Float16* __restrict__ A,
                                                 const _Float16* __restrict__ BT,
                                                 void* __restrict__ Cout,
                                                 int M, int seqN) {
  __shared__ __align__(64) _Float16 As[64][32];
  __shared__ __align__(64) _Float16 Bs[64][32];
  const int t = threadIdx.x;
  const int wv = t >> 5, lane = t & 31;
  const int hf = lane >> 4, l16 = lane & 15, kb8 = hf * 8;
  const int m0 = blockIdx.x * 64, n0 = blockIdx.y * 64;
  const int srow = t >> 1, shalf = (t & 1) * 16;

  v8f acc[4] = {};
  for (int k0 = 0; k0 < 512; k0 += 32) {
    __syncthreads();
    cp_async_b128(&A [(size_t)(m0 + srow) * 512 + k0 + shalf],     &As[srow][shalf]);
    cp_async_b128(&A [(size_t)(m0 + srow) * 512 + k0 + shalf + 8], &As[srow][shalf + 8]);
    cp_async_b128(&BT[(size_t)(n0 + srow) * 512 + k0 + shalf],     &Bs[srow][shalf]);
    cp_async_b128(&BT[(size_t)(n0 + srow) * 512 + k0 + shalf + 8], &Bs[srow][shalf + 8]);
    cp_async_wait();
    __syncthreads();
    // A fragment: lane holds row (wv*16+l16), a[i] = A[row][k = hf*16+i]
    v8h alo = *(const v8h*)&As[wv * 16 + l16][hf * 16];
    v8h ahi = *(const v8h*)&As[wv * 16 + l16][hf * 16 + 8];
    v16h af = CAT16(alo, ahi);
#pragma unroll
    for (int tt = 0; tt < 4; ++tt) {
      // B fragment: b[i] = B[k = hf*16+i][n = tt*16+l16] = Bs[tt*16+l16][hf*16+i]
      v16h bf = *(const v16h*)&Bs[tt * 16 + l16][hf * 16];
      acc[tt] = wmma_f16(af, bf, acc[tt]);
    }
  }
#pragma unroll
  for (int tt = 0; tt < 4; ++tt) {
#pragma unroll
    for (int r = 0; r < 8; ++r) {
      int m = m0 + wv * 16 + r + hf * 8;
      int c = n0 + tt * 16 + l16;
      float v = acc[tt][r];
      if (MODE == 0) {
        ((float*)Cout)[(size_t)m * 512 + c] = v;
      } else if (MODE == 1) {
        int b = m / seqN, n = m % seqN, h = c >> 6, d = c & 63;
        ((_Float16*)Cout)[(((size_t)(b * NH + h) * seqN + n) << 6) + d] = (_Float16)v;
      } else {
        int b = m / seqN, kk = m % seqN, h = c >> 6, d = c & 63;
        ((_Float16*)Cout)[((size_t)(b * NH + h) * HD + d) * seqN + kk] = (_Float16)v;
      }
    }
  }
}

// ---------------------------------------------------------------- bias MLP (WMMA layer-2)
// bias[b][h][q][k] = silu(d(q,k)*W1 + b1) @ W2 + b2
__global__ __launch_bounds__(128) void bias_kernel(const float* __restrict__ qc,
                                                   const float* __restrict__ kc,
                                                   const float* __restrict__ W1,
                                                   const float* __restrict__ b1,
                                                   const float* __restrict__ W2,
                                                   const float* __restrict__ b2,
                                                   _Float16* __restrict__ biasOut) {
  __shared__ float sW1[64], sb1[64], sW2[512], sb2[8];
  const int t = threadIdx.x;
  if (t < 64) { sW1[t] = W1[t]; sb1[t] = b1[t]; }
  for (int i = t; i < 512; i += 128) sW2[i] = W2[i];
  if (t < 8) sb2[t] = b2[t];
  __syncthreads();

  const int wv = t >> 5, lane = t & 31;
  const int hf = lane >> 4, l16 = lane & 15, kb8 = hf * 8;
  const int unit = blockIdx.x * 4 + wv;           // (b, q, 128-key block)
  const int kblk = unit & 15;
  const int q = (unit >> 4) & (NQ - 1);
  const int b = unit >> 14;

  // B fragments from W2 (64x8, cols 8..15 zero-padded), split hidden dim into 2 chunks
  v16h bw0, bw1;
#pragma unroll
  for (int i = 0; i < 16; ++i) {
    int j0 = kb8 + ((i >> 3) << 4) + (i & 7);
    bw0[i] = (l16 < 8) ? (_Float16)sW2[j0 * 8 + l16] : (_Float16)0.0f;
    bw1[i] = (l16 < 8) ? (_Float16)sW2[(32 + j0) * 8 + l16] : (_Float16)0.0f;
  }
  v8f cinit;
  {
    float cb = (l16 < 8) ? sb2[l16] : 0.0f;
#pragma unroll
    for (int r = 0; r < 8; ++r) cinit[r] = cb;
  }
  const float qx = qc[(size_t)(b * NQ + q) * 3 + 0];
  const float qy = qc[(size_t)(b * NQ + q) * 3 + 1];
  const float qz = qc[(size_t)(b * NQ + q) * 3 + 2];

  for (int it = 0; it < 8; ++it) {
    const int k0 = kblk * 128 + it * 16;
    const int kk = k0 + l16;                       // A-row = this lane's pair row
    float dx = qx - kc[(size_t)(b * NK + kk) * 3 + 0];
    float dy = qy - kc[(size_t)(b * NK + kk) * 3 + 1];
    float dz = qz - kc[(size_t)(b * NK + kk) * 3 + 2];
    float d = sqrtf(dx * dx + dy * dy + dz * dz);
    v16h a0, a1;
#pragma unroll
    for (int i = 0; i < 16; ++i) {
      int jj = kb8 + ((i >> 3) << 4) + (i & 7);    // ISA A-layout slot -> hidden idx
      a0[i] = (_Float16)fast_silu(d * sW1[jj] + sb1[jj]);
      a1[i] = (_Float16)fast_silu(d * sW1[32 + jj] + sb1[32 + jj]);
    }
    v8f dacc = wmma_f16(a1, bw1, cinit);
    dacc = wmma_f16(a0, bw0, dacc);
    if (l16 < 8) {                                  // lane's column = head index
      v8h pk;
#pragma unroll
      for (int r = 0; r < 8; ++r) pk[r] = (_Float16)dacc[r];
      *(v8h*)&biasOut[((size_t)(b * NH + l16) * NQ + q) * NK + k0 + hf * 8] = pk;
    }
  }
}

// ---------------------------------------------------------------- flash attention
__global__ __launch_bounds__(128) void attn_kernel(const _Float16* __restrict__ Q,
                                                   const _Float16* __restrict__ K,
                                                   const _Float16* __restrict__ VT,
                                                   const _Float16* __restrict__ Bias,
                                                   _Float16* __restrict__ O16) {
  __shared__ __align__(64) _Float16 Sp[4][16][32];  // per-wave P staging
  const int t = threadIdx.x;
  const int wv = t >> 5, lane = t & 31;
  const int hf = lane >> 4, l16 = lane & 15, kb8 = hf * 8;
  const int unit = blockIdx.x * 4 + wv;             // (b, h, q-tile)
  const int qt = unit & 63;
  const int h = (unit >> 6) & 7;
  const int b = unit >> 9;
  const int q0 = qt * 16;
  const size_t Qbase = ((size_t)(b * NH + h) * NQ + q0) * HD;
  const size_t Kbase = (size_t)(b * NH + h) * NK * HD;
  const size_t Vbase = (size_t)(b * NH + h) * HD * NK;
  const size_t Bbase = ((size_t)(b * NH + h) * NQ + q0) * NK;

  v16h aq0, aq1;                                    // Q fragment (scale baked into Wq)
  {
    v8h a0 = *(const v8h*)&Q[Qbase + (size_t)l16 * HD + hf * 16];
    v8h a1 = *(const v8h*)&Q[Qbase + (size_t)l16 * HD + hf * 16 + 8];
    v8h a2 = *(const v8h*)&Q[Qbase + (size_t)l16 * HD + 32 + hf * 16];
    v8h a3 = *(const v8h*)&Q[Qbase + (size_t)l16 * HD + 40 + hf * 16];
    aq0 = CAT16(a0, a1);
    aq1 = CAT16(a2, a3);
  }
  float mrow[8], lrow[8];
  v8f oacc[4] = {};
#pragma unroll
  for (int r = 0; r < 8; ++r) { mrow[r] = -INFINITY; lrow[r] = 0.0f; }

  for (int kt = 0; kt < NK / 32; ++kt) {
    const int k0 = kt * 32;
    // prefetch next tile of K/V/bias into cache (global_prefetch_b8)
    if (kt + 1 < NK / 32) {
      __builtin_prefetch(&K[Kbase + (size_t)(k0 + 32 + lane) * HD], 0, 3);
      __builtin_prefetch(&VT[Vbase + (size_t)lane * NK + k0 + 32], 0, 3);
      __builtin_prefetch(&VT[Vbase + (size_t)(32 + lane) * NK + k0 + 32], 0, 3);
      __builtin_prefetch(&Bias[Bbase + (size_t)l16 * NK + k0 + 32], 0, 3);
    }
    v8f S[2];
#pragma unroll
    for (int sub = 0; sub < 2; ++sub) {
      v8f c;
#pragma unroll
      for (int r = 0; r < 8; ++r)
        c[r] = (float)Bias[Bbase + (size_t)(r + hf * 8) * NK + k0 + sub * 16 + l16];
      v16h bk0 = *(const v16h*)&K[Kbase + (size_t)(k0 + sub * 16 + l16) * HD + hf * 16];
      v16h bk1 = *(const v16h*)&K[Kbase + (size_t)(k0 + sub * 16 + l16) * HD + 32 + hf * 16];
      c = wmma_f16(aq1, bk1, c);
      S[sub] = wmma_f16(aq0, bk0, c);
    }
    // online softmax per row (rows live on 16-lane halves in C/D layout)
#pragma unroll
    for (int r = 0; r < 8; ++r) {
      float mx = fmaxf(S[0][r], S[1][r]);
      mx = fmaxf(mx, __shfl_xor(mx, 1, 32));
      mx = fmaxf(mx, __shfl_xor(mx, 2, 32));
      mx = fmaxf(mx, __shfl_xor(mx, 4, 32));
      mx = fmaxf(mx, __shfl_xor(mx, 8, 32));
      float mnew = fmaxf(mrow[r], mx);
      float corr = __expf(mrow[r] - mnew);
      float p0 = __expf(S[0][r] - mnew);
      float p1 = __expf(S[1][r] - mnew);
      S[0][r] = p0; S[1][r] = p1;
      float rs = p0 + p1;
      rs += __shfl_xor(rs, 1, 32);
      rs += __shfl_xor(rs, 2, 32);
      rs += __shfl_xor(rs, 4, 32);
      rs += __shfl_xor(rs, 8, 32);
      lrow[r] = lrow[r] * corr + rs;
      mrow[r] = mnew;
#pragma unroll
      for (int ch = 0; ch < 4; ++ch) oacc[ch][r] *= corr;
    }
    // D-layout -> A-layout transpose through wave-private LDS
#pragma unroll
    for (int r = 0; r < 8; ++r) {
      Sp[wv][r + hf * 8][l16]      = (_Float16)S[0][r];
      Sp[wv][r + hf * 8][16 + l16] = (_Float16)S[1][r];
    }
    v8h plo = *(const v8h*)&Sp[wv][l16][hf * 16];
    v8h phi = *(const v8h*)&Sp[wv][l16][hf * 16 + 8];
    v16h ap = CAT16(plo, phi);
#pragma unroll
    for (int ch = 0; ch < 4; ++ch) {
      v16h bv = *(const v16h*)&VT[Vbase + (size_t)(ch * 16 + l16) * NK + k0 + hf * 16];
      oacc[ch] = wmma_f16(ap, bv, oacc[ch]);
    }
  }
#pragma unroll
  for (int r = 0; r < 8; ++r) {
    float inv = fast_rcp(lrow[r]);
    int qrow = q0 + r + hf * 8;
#pragma unroll
    for (int ch = 0; ch < 4; ++ch)
      O16[(size_t)(b * NQ + qrow) * HIDX + h * HD + ch * 16 + l16] =
          (_Float16)(oacc[ch][r] * inv);
  }
}

// ---------------------------------------------------------------- launcher
extern "C" void kernel_launch(void* const* d_in, const int* in_sizes, int n_in,
                              void* d_out, int out_size, void* d_ws, size_t ws_size,
                              hipStream_t stream) {
  const float* q_in      = (const float*)d_in[0];
  const float* kv_in     = (const float*)d_in[1];
  const float* q_coords  = (const float*)d_in[2];
  const float* kv_coords = (const float*)d_in[3];
  const float* Wq = (const float*)d_in[4];
  const float* Wk = (const float*)d_in[5];
  const float* Wv = (const float*)d_in[6];
  const float* Wo = (const float*)d_in[7];
  const float* W1 = (const float*)d_in[8];
  const float* b1 = (const float*)d_in[9];
  const float* W2 = (const float*)d_in[10];
  const float* b2 = (const float*)d_in[11];

  char* ws = (char*)d_ws;
  _Float16* q16   = (_Float16*)(ws);                              // 2 MB
  _Float16* kv16  = (_Float16*)(ws + (2ull << 20));               // 4 MB
  _Float16* WqT   = (_Float16*)(ws + (6ull << 20));               // 512 KB each
  _Float16* WkT   = (_Float16*)(ws + (6ull << 20) + 1 * 524288);
  _Float16* WvT   = (_Float16*)(ws + (6ull << 20) + 2 * 524288);
  _Float16* WoT   = (_Float16*)(ws + (6ull << 20) + 3 * 524288);
  _Float16* Q16   = (_Float16*)(ws + (8ull << 20));               // 2 MB  [B][H][Nq][64]
  _Float16* K16   = (_Float16*)(ws + (10ull << 20));              // 4 MB  [B][H][Nk][64]
  _Float16* VT16  = (_Float16*)(ws + (14ull << 20));              // 4 MB  [B][H][64][Nk]
  _Float16* O16   = (_Float16*)(ws + (18ull << 20));              // 2 MB
  _Float16* BiasT = (_Float16*)(ws + (20ull << 20));              // 64 MB [B][H][Nq][Nk]

  cvt_act<<<(BB * NQ * DIMX) / 256, 256, 0, stream>>>(q_in, q16, BB * NQ * DIMX);
  cvt_act<<<(BB * NK * DIMX) / 256, 256, 0, stream>>>(kv_in, kv16, BB * NK * DIMX);
  cvt_wT<<<1024, 256, 0, stream>>>(Wq, WqT, 0.125f);  // fold 1/sqrt(64) into Wq
  cvt_wT<<<1024, 256, 0, stream>>>(Wk, WkT, 1.0f);
  cvt_wT<<<1024, 256, 0, stream>>>(Wv, WvT, 1.0f);
  cvt_wT<<<1024, 256, 0, stream>>>(Wo, WoT, 1.0f);

  gemm_f16k<1><<<dim3(BB * NQ / 64, 8), 128, 0, stream>>>(q16,  WqT, Q16,  BB * NQ, NQ);
  gemm_f16k<1><<<dim3(BB * NK / 64, 8), 128, 0, stream>>>(kv16, WkT, K16,  BB * NK, NK);
  gemm_f16k<2><<<dim3(BB * NK / 64, 8), 128, 0, stream>>>(kv16, WvT, VT16, BB * NK, NK);

  bias_kernel<<<(BB * NQ * 16) / 4, 128, 0, stream>>>(q_coords, kv_coords, W1, b1, W2, b2, BiasT);
  attn_kernel<<<(BB * NH * (NQ / 16)) / 4, 128, 0, stream>>>(Q16, K16, VT16, BiasT, O16);

  gemm_f16k<0><<<dim3(BB * NQ / 64, 8), 128, 0, stream>>>(O16, WoT, d_out, BB * NQ, NQ);
}
